// BidirectionalScanStrategy_7662221656540
// MI455X (gfx1250) — matrix operand order repaired and checked by reading, not compiled, not run on hardware
//
#include <hip/hip_runtime.h>
#include <hip/hip_bf16.h>
#include <math.h>

#define DMODEL 192
#define DINNER 384
#define DSTATE 16
#define DTRANK 12
#define LSEQ   4096
#define NBATCH 2
#define BL     (NBATCH*LSEQ)   /* 8192 */
#define N1     (2*DINNER)      /* 768: [xp | z] */
#define N2PAD  448             /* 16(B)+16(C)+384(dt) padded to 448 */
#define NOUT   DMODEL          /* 192 */
#define NCHUNK 64
#define CLEN   64              /* NCHUNK*CLEN == LSEQ */
#define NBD    (NBATCH*DINNER) /* 768 */

#if defined(__has_builtin)
#if __has_builtin(__builtin_amdgcn_global_load_async_to_lds_b128) && \
    __has_builtin(__builtin_amdgcn_s_wait_asynccnt)
#define HAVE_ASYNC_LDS 1
#endif
#endif

typedef __attribute__((ext_vector_type(16))) __bf16 v16bf;
typedef __attribute__((ext_vector_type(8)))  __bf16 v8bf;
typedef __attribute__((ext_vector_type(8)))  float  v8f;
typedef __attribute__((ext_vector_type(4)))  float  v4f;
typedef __attribute__((ext_vector_type(4)))  int    v4i;
typedef v4i __attribute__((address_space(1)))* v4i_gptr;   // global (AS1)
typedef v4i __attribute__((address_space(3)))* v4i_lptr;   // LDS (AS3)

// ---------------- WMMA fragment loads (bf16 16x16x32, wave32 layouts) --------

__device__ __forceinline__ v16bf load_a_frag(const __bf16* __restrict__ A, int lda,
                                             int m0, int k0, int lane) {
  // 16-bit A 16x32: lane<16 holds row=lane, K k0..k0+15;
  //                 lane>=16 holds row=lane-16, K k0+16..k0+31
  int half = lane >> 4, row = lane & 15;
  const __bf16* p = A + (size_t)(m0 + row) * lda + k0 + half * 16;
  v8bf lo = *(const v8bf*)(p);
  v8bf hi = *(const v8bf*)(p + 8);
  return __builtin_shufflevector(lo, hi, 0,1,2,3,4,5,6,7,8,9,10,11,12,13,14,15);
}

__device__ __forceinline__ v16bf load_b_frag(const __bf16* __restrict__ Bt, int ldb,
                                             int n0, int k0, int lane) {
  // B 32x16: lanes 0-15 hold K=k0..k0+15 of col n0+lane; lanes 16-31 K=k0+16..31.
  // Bt is (N x K) row-major so this is one contiguous 32B load.
  int half = lane >> 4, col = lane & 15;
  const __bf16* p = Bt + (size_t)(n0 + col) * ldb + k0 + half * 16;
  return *(const v16bf*)p;
}

// ---------------- generic bf16 WMMA GEMM: C[MxN] = A[MxK] @ Bt[NxK]^T --------
// mode 0: plain row-major f32 store (ld = Cld)
// mode 1: store  scale*acc into (B, DMODEL, LSEQ) layout (row r -> b,t; col -> c)
// mode 2: read-add scale*acc into same layout
__global__ __launch_bounds__(128)
void gemm_wmma_bf16(const __bf16* __restrict__ A, const __bf16* __restrict__ Bt,
                    float* __restrict__ C, int M, int N, int K,
                    int mode, float scale, int Cld) {
  const int lane = threadIdx.x & 31;
  const int wave = threadIdx.x >> 5;
  const int m0 = blockIdx.y * 64 + wave * 16;   // each wave: 16 rows x 64 cols
  const int n0 = blockIdx.x * 64;

  v8f acc[4] = {v8f{}, v8f{}, v8f{}, v8f{}};
  for (int k = 0; k < K; k += 32) {
    v16bf a  = load_a_frag(A, K, m0, k, lane);
    v16bf b0 = load_b_frag(Bt, K, n0 +  0, k, lane);
    v16bf b1 = load_b_frag(Bt, K, n0 + 16, k, lane);
    v16bf b2 = load_b_frag(Bt, K, n0 + 32, k, lane);
    v16bf b3 = load_b_frag(Bt, K, n0 + 48, k, lane);
    acc[0] = __builtin_amdgcn_wmma_f32_16x16x32_bf16(false, a, false, b0, (short)0, acc[0], false, false);
    acc[1] = __builtin_amdgcn_wmma_f32_16x16x32_bf16(false, a, false, b1, (short)0, acc[1], false, false);
    acc[2] = __builtin_amdgcn_wmma_f32_16x16x32_bf16(false, a, false, b2, (short)0, acc[2], false, false);
    acc[3] = __builtin_amdgcn_wmma_f32_16x16x32_bf16(false, a, false, b3, (short)0, acc[3], false, false);
  }

  const int crow = (lane >> 4) * 8;     // C layout: VGPR i -> row crow+i
  const int ccol = lane & 15;
  if (mode == 0) {
#pragma unroll
    for (int j = 0; j < 4; ++j)
#pragma unroll
      for (int i = 0; i < 8; ++i)
        C[(size_t)(m0 + crow + i) * Cld + (n0 + j * 16 + ccol)] = acc[j][i];
  } else {
#pragma unroll
    for (int j = 0; j < 4; ++j) {
      const int c = n0 + j * 16 + ccol;
#pragma unroll
      for (int i = 0; i < 8; ++i) {
        const int r = m0 + crow + i;
        const int bb = r >> 12;           // LSEQ = 4096
        const int t  = r & (LSEQ - 1);
        const size_t di = ((size_t)bb * DMODEL + c) * LSEQ + t;
        float v = scale * acc[j][i];
        if (mode == 2) v += C[di];
        C[di] = v;
      }
    }
  }
}

// ---------------- prep kernels ----------------------------------------------

__global__ void k_wt_in(const float* __restrict__ W_in, __bf16* __restrict__ wt) {
  int idx = blockIdx.x * blockDim.x + threadIdx.x;
  if (idx >= N1 * DMODEL) return;
  int n = idx / DMODEL, k = idx % DMODEL;
  wt[idx] = (__bf16)W_in[(size_t)k * N1 + n];
}

// wt2: (N2PAD x DINNER). cols 0-15: Bm proj, 16-31: Cm proj, 32-415: W_x[:,:12]@W_dt, rest 0
__global__ void k_wt2(const float* __restrict__ W_x, const float* __restrict__ W_dt,
                      __bf16* __restrict__ wt) {
  int idx = blockIdx.x * blockDim.x + threadIdx.x;
  if (idx >= N2PAD * DINNER) return;
  int n = idx / DINNER, k = idx % DINNER;
  float v = 0.f;
  if (n < DSTATE) {
    v = W_x[(size_t)k * (DTRANK + 2 * DSTATE) + DTRANK + n];
  } else if (n < 2 * DSTATE) {
    v = W_x[(size_t)k * (DTRANK + 2 * DSTATE) + DTRANK + DSTATE + (n - DSTATE)];
  } else if (n < 2 * DSTATE + DINNER) {
    int dcol = n - 2 * DSTATE;
#pragma unroll
    for (int r = 0; r < DTRANK; ++r)
      v += W_x[(size_t)k * (DTRANK + 2 * DSTATE) + r] * W_dt[(size_t)r * DINNER + dcol];
  }
  wt[idx] = (__bf16)v;
}

__global__ void k_wt_out(const float* __restrict__ W_out, __bf16* __restrict__ wt) {
  int idx = blockIdx.x * blockDim.x + threadIdx.x;
  if (idx >= NOUT * DINNER) return;
  int n = idx / DINNER, k = idx % DINNER;
  wt[idx] = (__bf16)W_out[(size_t)k * NOUT + n];
}

// features (B, DMODEL, L) -> flat bf16 (B*L, DMODEL)
__global__ void k_flatten(const float* __restrict__ f, __bf16* __restrict__ flat) {
  int idx = blockIdx.x * blockDim.x + threadIdx.x;
  if (idx >= BL * DMODEL) return;
  int r = idx / DMODEL, c = idx % DMODEL;
  int b = r >> 12, t = r & (LSEQ - 1);
  flat[idx] = (__bf16)f[((size_t)b * DMODEL + c) * LSEQ + t];
}

__global__ void k_scalar(float* __restrict__ p, float v) { p[0] = v; }

// depthwise causal/anticausal conv over xp (= xz cols 0..383) + bias + silu
__global__ void k_conv(const float* __restrict__ xz, const float* __restrict__ cw,
                       const float* __restrict__ cb, float* __restrict__ xaf,
                       __bf16* __restrict__ xabf, int dir) {
  int idx = blockIdx.x * blockDim.x + threadIdx.x;
  if (idx >= BL * DINNER) return;
  int r = idx / DINNER, d = idx % DINNER;
  int b = r >> 12, t = r & (LSEQ - 1);
  float acc = cb[d];
#pragma unroll
  for (int k = 0; k < 4; ++k) {
    int tt = dir ? (t + 3 - k) : (t - 3 + k);
    if (tt >= 0 && tt < LSEQ)
      acc += cw[d * 4 + k] * xz[((size_t)b * LSEQ + tt) * N1 + d];
  }
  float s = acc / (1.0f + __expf(-acc));   // silu
  xaf[idx] = s;
  xabf[idx] = (__bf16)s;
}

// ---------------- chunked parallel selective scan ----------------------------
// recurrence h_t = a_t h_{t-1} + b_t, a_t = exp(dt*A[d,n]), b_t = dt*xa*Bm[t,n]
// phase1: per chunk, local scan from 0 -> (prod a, h_local_end)
// phase2: sequential prefix over chunk aggregates -> incoming state per chunk
// phase3: re-run chunk from true incoming state, emit gated y (bf16)

__device__ __forceinline__ float softplus_f(float x) {
  return (x > 20.f) ? x : log1pf(__expf(x));
}

// stage this chunk's [Bm|Cm] rows (CLEN x 32 f32 = 8KB) into LDS.
// Uses the gfx1250 async-to-LDS engine when the toolchain exposes it.
__device__ __forceinline__ void stage_chunk(float* sB, const float* __restrict__ xdbl,
                                            int b, int c, int dir) {
  for (int e = threadIdx.x; e < CLEN * 8; e += 128) {
    int i = e >> 3, j = (e & 7) << 2;          // row i, 4-float column group j
    int s = c * CLEN + i;
    int t = dir ? (LSEQ - 1 - s) : s;
    const float* src = xdbl + ((size_t)b * LSEQ + t) * N2PAD + j;
    float* dst = sB + i * 32 + j;
#if defined(HAVE_ASYNC_LDS)
    __builtin_amdgcn_global_load_async_to_lds_b128((v4i_gptr)src, (v4i_lptr)dst, 0, 0);
#else
    *(v4f*)dst = *(const v4f*)src;
#endif
  }
#if defined(HAVE_ASYNC_LDS)
  __builtin_amdgcn_s_wait_asynccnt(0);
#endif
  __syncthreads();
}

__global__ __launch_bounds__(128)
void k_scan_phase1(const float* __restrict__ xdbl, const float* __restrict__ xaf,
                   const float* __restrict__ A_log, const float* __restrict__ b_dt,
                   float* __restrict__ Aprod, float* __restrict__ Bagg, int dir) {
  __shared__ alignas(16) float sB[CLEN * 32];
  const int c   = blockIdx.x / 6;
  const int sub = blockIdx.x % 6;
  const int bd  = sub * 128 + threadIdx.x;   // 0..767, one (batch,channel) per lane
  const int b   = bd / DINNER, d = bd % DINNER;

  stage_chunk(sB, xdbl, b, c, dir);

  float Arow[DSTATE];
#pragma unroll
  for (int n = 0; n < DSTATE; ++n) Arow[n] = -__expf(A_log[(size_t)d * DSTATE + n]);
  const float bdt = b_dt[d];

  float ap[DSTATE], hl[DSTATE];
#pragma unroll
  for (int n = 0; n < DSTATE; ++n) { ap[n] = 1.f; hl[n] = 0.f; }

  for (int i = 0; i < CLEN; ++i) {
    int s = c * CLEN + i;
    int t = dir ? (LSEQ - 1 - s) : s;
    size_t r = (size_t)b * LSEQ + t;
    if (i + 4 < CLEN) {
      int t2 = dir ? (t - 4) : (t + 4);
      __builtin_prefetch(xdbl + ((size_t)b * LSEQ + t2) * N2PAD + 2 * DSTATE + d, 0, 0);
    }
    float dtv = softplus_f(xdbl[r * N2PAD + 2 * DSTATE + d] + bdt);
    float dbx = dtv * xaf[r * DINNER + d];
#pragma unroll
    for (int n = 0; n < DSTATE; ++n) {
      float a = __expf(dtv * Arow[n]);
      hl[n] = a * hl[n] + dbx * sB[i * 32 + n];
      ap[n] *= a;
    }
  }
  size_t o = ((size_t)c * NBD + bd) * DSTATE;
#pragma unroll
  for (int n = 0; n < DSTATE; ++n) { Aprod[o + n] = ap[n]; Bagg[o + n] = hl[n]; }
}

__global__ void k_scan_phase2(const float* __restrict__ Aprod,
                              const float* __restrict__ Bagg,
                              float* __restrict__ Hin) {
  int idx = blockIdx.x * blockDim.x + threadIdx.x;   // (bd, n) pair
  if (idx >= NBD * DSTATE) return;
  float h = 0.f;
  for (int c = 0; c < NCHUNK; ++c) {
    size_t o = (size_t)c * (NBD * DSTATE) + idx;
    Hin[o] = h;
    h = Bagg[o] + Aprod[o] * h;
  }
}

__global__ __launch_bounds__(128)
void k_scan_phase3(const float* __restrict__ xdbl, const float* __restrict__ xaf,
                   const float* __restrict__ xz, const float* __restrict__ A_log,
                   const float* __restrict__ Dv, const float* __restrict__ b_dt,
                   const float* __restrict__ Hin, __bf16* __restrict__ ybf, int dir) {
  __shared__ alignas(16) float sB[CLEN * 32];
  const int c   = blockIdx.x / 6;
  const int sub = blockIdx.x % 6;
  const int bd  = sub * 128 + threadIdx.x;
  const int b   = bd / DINNER, d = bd % DINNER;

  stage_chunk(sB, xdbl, b, c, dir);

  float Arow[DSTATE];
#pragma unroll
  for (int n = 0; n < DSTATE; ++n) Arow[n] = -__expf(A_log[(size_t)d * DSTATE + n]);
  const float bdt = b_dt[d];
  const float Dd  = Dv[d];

  float h[DSTATE];
  {
    size_t o = ((size_t)c * NBD + bd) * DSTATE;
#pragma unroll
    for (int n = 0; n < DSTATE; ++n) h[n] = Hin[o + n];
  }

  for (int i = 0; i < CLEN; ++i) {
    int s = c * CLEN + i;
    int t = dir ? (LSEQ - 1 - s) : s;
    size_t r = (size_t)b * LSEQ + t;
    if (i + 4 < CLEN) {
      int t2 = dir ? (t - 4) : (t + 4);
      __builtin_prefetch(xdbl + ((size_t)b * LSEQ + t2) * N2PAD + 2 * DSTATE + d, 0, 0);
    }
    float dtv = softplus_f(xdbl[r * N2PAD + 2 * DSTATE + d] + bdt);
    float xav = xaf[r * DINNER + d];
    float dbx = dtv * xav;
    float y = 0.f;
#pragma unroll
    for (int n = 0; n < DSTATE; ++n) {
      float a = __expf(dtv * Arow[n]);
      h[n] = a * h[n] + dbx * sB[i * 32 + n];
      y += h[n] * sB[i * 32 + 16 + n];
    }
    float zv = xz[r * N1 + DINNER + d];
    float sz = zv / (1.0f + __expf(-zv));
    ybf[r * DINNER + d] = (__bf16)((y + Dd * xav) * sz);
  }
}

// ---------------- host launcher ---------------------------------------------

extern "C" void kernel_launch(void* const* d_in, const int* in_sizes, int n_in,
                              void* d_out, int out_size, void* d_ws, size_t ws_size,
                              hipStream_t stream) {
  (void)in_sizes; (void)n_in; (void)ws_size;
  const float* features = (const float*)d_in[0];
  const float* W_in  = (const float*)d_in[1];
  const float* cw    = (const float*)d_in[2];
  const float* cb    = (const float*)d_in[3];
  const float* W_x   = (const float*)d_in[4];
  const float* W_dt  = (const float*)d_in[5];
  const float* bdt   = (const float*)d_in[6];
  const float* A_log = (const float*)d_in[7];
  const float* Dv    = (const float*)d_in[8];
  const float* W_out = (const float*)d_in[9];
  float* out = (float*)d_out;

  char* ws = (char*)d_ws;
  size_t off = 0;
  auto take = [&](size_t bytes) -> char* {
    char* p = ws + off;
    off = (off + bytes + 255) & ~(size_t)255;
    return p;
  };
  __bf16* wtin  = (__bf16*)take((size_t)N1 * DMODEL * 2);
  __bf16* wt2   = (__bf16*)take((size_t)N2PAD * DINNER * 2);
  __bf16* wtout = (__bf16*)take((size_t)NOUT * DINNER * 2);
  __bf16* flat  = (__bf16*)take((size_t)BL * DMODEL * 2);
  float*  xz    = (float*) take((size_t)BL * N1 * 4);
  float*  xaf   = (float*) take((size_t)BL * DINNER * 4);
  __bf16* xabf  = (__bf16*)take((size_t)BL * DINNER * 2);
  float*  xdbl  = (float*) take((size_t)BL * N2PAD * 4);
  __bf16* ybf   = (__bf16*)take((size_t)BL * DINNER * 2);
  float*  Aprod = (float*) take((size_t)NCHUNK * NBD * DSTATE * 4);
  float*  Bagg  = (float*) take((size_t)NCHUNK * NBD * DSTATE * 4);
  float*  Hin   = (float*) take((size_t)NCHUNK * NBD * DSTATE * 4);

  k_wt_in <<<(N1 * DMODEL + 255) / 256, 256, 0, stream>>>(W_in, wtin);
  k_wt2   <<<(N2PAD * DINNER + 255) / 256, 256, 0, stream>>>(W_x, W_dt, wt2);
  k_wt_out<<<(NOUT * DINNER + 255) / 256, 256, 0, stream>>>(W_out, wtout);
  k_flatten<<<(BL * DMODEL + 255) / 256, 256, 0, stream>>>(features, flat);
  if (out_size > NBATCH * DMODEL * LSEQ)
    k_scalar<<<1, 1, 0, stream>>>(out + (size_t)NBATCH * DMODEL * LSEQ, (float)LSEQ);

  // GEMM1 (shared by both directions): xz = flat @ W_in
  gemm_wmma_bf16<<<dim3(N1 / 64, BL / 64), 128, 0, stream>>>(
      flat, wtin, xz, BL, N1, DMODEL, 0, 1.0f, N1);

  for (int dir = 0; dir < 2; ++dir) {
    k_conv<<<(BL * DINNER + 255) / 256, 256, 0, stream>>>(xz, cw, cb, xaf, xabf, dir);
    // GEMM2: [Bm | Cm | dt_in] = xa @ [W_x[:,12:44] | W_x[:,:12]@W_dt]
    gemm_wmma_bf16<<<dim3(N2PAD / 64, BL / 64), 128, 0, stream>>>(
        xabf, wt2, xdbl, BL, N2PAD, DINNER, 0, 1.0f, N2PAD);
    // chunked scan: 64x more parallelism than the naive serial scan
    k_scan_phase1<<<NCHUNK * 6, 128, 0, stream>>>(xdbl, xaf, A_log, bdt, Aprod, Bagg, dir);
    k_scan_phase2<<<(NBD * DSTATE + 255) / 256, 256, 0, stream>>>(Aprod, Bagg, Hin);
    k_scan_phase3<<<NCHUNK * 6, 128, 0, stream>>>(xdbl, xaf, xz, A_log, Dv, bdt, Hin, ybf, dir);
    // GEMM3: out(B,C,L) (+)= 0.5 * (y @ W_out)
    gemm_wmma_bf16<<<dim3(NOUT / 64, BL / 64), 128, 0, stream>>>(
        ybf, wtout, out, BL, NOUT, DINNER, dir == 0 ? 1 : 2, 0.5f, NOUT);
  }
}
